// StandardGNN_60962765799636
// MI455X (gfx1250) — matrix operand order repaired and checked by reading, not compiled, not run on hardware
//
#include <hip/hip_runtime.h>

// StandardGNN on MI455X (gfx1250).
// - GEMMs via v_wmma_f32_16x16x32_bf16 (bf16 A/B staged in LDS, f32 accumulate).
// - Message passing via f32 global atomics (agg buffer is L2-resident: 51MB << 192MB L2).
// - BatchNorm(train) + ReLU with two-level reduction for per-feature stats.

#define HID 128
#define BN_EPS 1e-5f

typedef __attribute__((ext_vector_type(16))) __bf16 v16bf;
typedef __attribute__((ext_vector_type(2)))  __bf16 bf16x2;
typedef __attribute__((ext_vector_type(8)))  float  v8f;

// ---------------- graph normalization ----------------

__global__ void gnn_init_deg(float* deg, int N) {
    int i = blockIdx.x * blockDim.x + threadIdx.x;
    if (i < N) deg[i] = 1.0f;   // self loop contributes 1 to every node's degree
}

__global__ void gnn_deg_edges(const int* __restrict__ dst, float* __restrict__ deg, int E) {
    int e = blockIdx.x * blockDim.x + threadIdx.x;
    if (e < E) atomicAdd(&deg[dst[e]], 1.0f);
}

__global__ void gnn_rsqrt_inplace(float* deg, int N) {
    int i = blockIdx.x * blockDim.x + threadIdx.x;
    if (i < N) deg[i] = rsqrtf(deg[i]);   // deg >= 1 always (self loop)
}

__global__ void gnn_edge_norm(const int* __restrict__ src, const int* __restrict__ dst,
                              const float* __restrict__ dinv, float* __restrict__ norm, int E) {
    int e = blockIdx.x * blockDim.x + threadIdx.x;
    if (e < E) norm[e] = dinv[src[e]] * dinv[dst[e]];
}

// ---------------- WMMA GEMM: C[N,Ncols] = A[N,128] @ W[128,Ncols] (+bias) ----------------
// One block = 16-row tile x all Ncols columns; wave w computes column tile w.
// A tile and W are staged in LDS as bf16 (W transposed so K-pairs are contiguous),
// inner loop is pure ds_load_b32 (packed bf16x2) + v_wmma_f32_16x16x32_bf16.

__global__ __launch_bounds__(256)
void gnn_gemm_wmma(const float* __restrict__ A, const float* __restrict__ W,
                   const float* __restrict__ bias, float* __restrict__ C,
                   int N, int Ncols) {
    __shared__ __bf16 sA[16 * HID];     // 4 KB : A tile, row-major [r][k]
    __shared__ __bf16 sWt[HID * HID];   // 32 KB: W transposed, [c][k]

    const int t    = threadIdx.x;
    const int row0 = blockIdx.x * 16;

    // stage A tile (16 x 128) -> bf16, coalesced
    for (int idx = t; idx < 16 * HID; idx += blockDim.x) {
        int row = row0 + (idx >> 7);
        if (row >= N) row = N - 1;                       // clamp (N=100000 is exact)
        sA[idx] = (__bf16)A[(size_t)row * HID + (idx & (HID - 1))];
    }
    // stage W (128 x Ncols) -> bf16 transposed, coalesced reads
    const int wcount = HID * Ncols;
    for (int idx = t; idx < wcount; idx += blockDim.x) {
        const int k = idx / Ncols;
        const int c = idx - k * Ncols;
        sWt[c * HID + k] = (__bf16)W[idx];
    }
    __syncthreads();

    const int lane = t & 31;
    const int wv   = t >> 5;           // wave id -> column tile
    const int m    = lane & 15;
    const int g    = lane >> 4;        // lane half
    const int col0 = wv * 16;

    v8f acc = {};
#pragma unroll
    for (int kb = 0; kb < HID; kb += 32) {
        v16bf a, b;
#pragma unroll
        for (int v = 0; v < 8; ++v) {
            // A 16x32 bf16 layout (ISA 7.12.2): VGPR v holds K pair starting at
            // (v&3)*2 + (v>>2)*16 + 8*g ; pair is contiguous -> one ds_load_b32
            const int ka = kb + ((v & 3) * 2 + (v >> 2) * 16 + g * 8);
            const bf16x2 ap = *(const bf16x2*)&sA[m * HID + ka];
            a[2 * v]     = ap.x;
            a[2 * v + 1] = ap.y;
            // B 32x16 bf16 layout: same K order as A (pair starting at
            // (v&3)*2 + (v>>2)*16 + 8*g); column = lane&15. Transposed LDS makes
            // the K pair contiguous -> one ds_load_b32.
            const int kw = kb + ((v & 3) * 2 + (v >> 2) * 16 + g * 8);
            const bf16x2 bp = *(const bf16x2*)&sWt[(col0 + m) * HID + kw];
            b[2 * v]     = bp.x;
            b[2 * v + 1] = bp.y;
        }
        acc = __builtin_amdgcn_wmma_f32_16x16x32_bf16(
            /*neg_a=*/false, a, /*neg_b=*/false, b,
            /*c_mod=*/(short)0, acc, /*reuse_a=*/false, /*reuse_b=*/false);
    }

    const float badd = bias ? bias[col0 + m] : 0.0f;
#pragma unroll
    for (int r = 0; r < 8; ++r) {
        // C/D layout: VGPR r -> row r (lanes 0-15) / row r+8 (lanes 16-31), col = lane&15
        const int row = row0 + r + g * 8;
        if (row < N) C[(size_t)row * Ncols + col0 + m] = acc[r] + badd;
    }
}

// ---------------- message passing ----------------

// agg[i,f] = bias[f] + h[i,f] * dinv[i]^2   (self-loop term + bias, also zero-inits agg)
__global__ void gnn_agg_init(const float* __restrict__ h, const float* __restrict__ dinv,
                             const float* __restrict__ bias, float* __restrict__ agg, int N) {
    size_t i = (size_t)blockIdx.x * blockDim.x + threadIdx.x;
    if (i >= (size_t)N * HID) return;
    const int node = (int)(i >> 7);
    const int f    = (int)(i & (HID - 1));
    const float d  = dinv[node];
    agg[i] = bias[f] + h[i] * d * d;
}

// agg[dst] += h[src] * norm  : 32 threads per edge, float4 per thread, f32 atomics (L2)
__global__ void gnn_scatter_edges(const float* __restrict__ h, const int* __restrict__ src,
                                  const int* __restrict__ dst, const float* __restrict__ norm,
                                  float* __restrict__ agg, int E) {
    size_t tid = (size_t)blockIdx.x * blockDim.x + threadIdx.x;
    const int e = (int)(tid >> 5);
    if (e >= E) return;
    const int c = ((int)tid & 31) * 4;
    const float nrm = norm[e];
    const float4 v = *(const float4*)(h + (size_t)src[e] * HID + c);
    float* o = agg + (size_t)dst[e] * HID + c;
    atomicAdd(o + 0, v.x * nrm);
    atomicAdd(o + 1, v.y * nrm);
    atomicAdd(o + 2, v.z * nrm);
    atomicAdd(o + 3, v.w * nrm);
}

// ---------------- batch norm (training stats) + ReLU ----------------

__global__ void gnn_zero_stats(float* stats) { stats[threadIdx.x] = 0.0f; }  // <<<1,256>>>

// stats[0..127] = sum_f, stats[128..255] = sumsq_f ; two-level: LDS then atomics
__global__ __launch_bounds__(256)
void gnn_bn_stats(const float* __restrict__ x, float* __restrict__ stats, int N) {
    __shared__ float s1[256], s2[256];
    const int f    = threadIdx.x & (HID - 1);
    const int half = threadIdx.x >> 7;           // 0 or 1: two rows per block-iteration
    float a = 0.0f, b = 0.0f;
    for (int i = blockIdx.x * 2 + half; i < N; i += gridDim.x * 2) {
        const float v = x[(size_t)i * HID + f];
        a += v;
        b += v * v;
    }
    s1[threadIdx.x] = a;
    s2[threadIdx.x] = b;
    __syncthreads();
    if (threadIdx.x < HID) {
        atomicAdd(&stats[f],       s1[f] + s1[f + HID]);
        atomicAdd(&stats[HID + f], s2[f] + s2[f + HID]);
    }
}

__global__ void gnn_bn_relu(float* __restrict__ x, const float* __restrict__ gamma,
                            const float* __restrict__ beta, const float* __restrict__ stats,
                            int N) {
    size_t i = (size_t)blockIdx.x * blockDim.x + threadIdx.x;
    if (i >= (size_t)N * HID) return;
    const int f = (int)(i & (HID - 1));
    const float invN = 1.0f / (float)N;
    const float mean = stats[f] * invN;
    const float var  = stats[HID + f] * invN - mean * mean;
    const float y = (x[i] - mean) * rsqrtf(var + BN_EPS) * gamma[f] + beta[f];
    x[i] = y > 0.0f ? y : 0.0f;
}

// ---------------- launch ----------------

extern "C" void kernel_launch(void* const* d_in, const int* in_sizes, int n_in,
                              void* d_out, int out_size, void* d_ws, size_t ws_size,
                              hipStream_t stream) {
    const float* x  = (const float*)d_in[0];
    const int*   ei = (const int*)d_in[1];
    const int N = in_sizes[0] / HID;
    const int E = in_sizes[1] / 2;
    const int* src = ei;         // edge_index[0]
    const int* dst = ei + E;     // edge_index[1]

    const float* W[3]   = {(const float*)d_in[2],  (const float*)d_in[6],  (const float*)d_in[10]};
    const float* bi[3]  = {(const float*)d_in[3],  (const float*)d_in[7],  (const float*)d_in[11]};
    const float* gam[3] = {(const float*)d_in[4],  (const float*)d_in[8],  (const float*)d_in[12]};
    const float* bet[3] = {(const float*)d_in[5],  (const float*)d_in[9],  (const float*)d_in[13]};
    const float* Wr = (const float*)d_in[14];
    const float* br = (const float*)d_in[15];
    const int OUT = in_sizes[15];            // 64

    // workspace layout (all re-initialized every call)
    float* ws    = (float*)d_ws;
    float* dinv  = ws;                       ws += N;
    float* norm  = ws;                       ws += E;
    float* stats = ws;                       ws += 256;
    float* bufA  = ws;                       ws += (size_t)N * HID;   // GEMM output h
    float* bufB  = ws;                                               // agg / layer output

    const dim3 b256(256);
    const int nhBlocks = (int)(((size_t)N * HID + 255) / 256);

    // symmetric GCN normalization
    gnn_init_deg<<<(N + 255) / 256, b256, 0, stream>>>(dinv, N);
    gnn_deg_edges<<<(E + 255) / 256, b256, 0, stream>>>(dst, dinv, E);
    gnn_rsqrt_inplace<<<(N + 255) / 256, b256, 0, stream>>>(dinv, N);
    gnn_edge_norm<<<(E + 255) / 256, b256, 0, stream>>>(src, dst, dinv, norm, E);

    const float* in = x;
    for (int l = 0; l < 3; ++l) {
        gnn_gemm_wmma<<<dim3((N + 15) / 16), dim3(32 * (HID / 16)), 0, stream>>>(
            in, W[l], nullptr, bufA, N, HID);
        gnn_agg_init<<<nhBlocks, b256, 0, stream>>>(bufA, dinv, bi[l], bufB, N);
        gnn_scatter_edges<<<(int)(((size_t)E * 32 + 255) / 256), b256, 0, stream>>>(
            bufA, src, dst, norm, bufB, E);
        gnn_zero_stats<<<1, 256, 0, stream>>>(stats);
        gnn_bn_stats<<<256, b256, 0, stream>>>(bufB, stats, N);
        gnn_bn_relu<<<nhBlocks, b256, 0, stream>>>(bufB, gam[l], bet[l], stats, N);
        in = bufB;
    }

    // readout: [N,128] @ [128,64] + br -> d_out (4 waves = 4 column tiles)
    gnn_gemm_wmma<<<dim3((N + 15) / 16), dim3(32 * (OUT / 16)), 0, stream>>>(
        in, Wr, br, (float*)d_out, N, OUT);
}
